// VGSSM_40492951666853
// MI455X (gfx1250) — matrix operand order repaired and checked by reading, not compiled, not run on hardware
//
#include <hip/hip_runtime.h>
#include <hip/hip_bf16.h>
#include <math.h>

typedef __attribute__((ext_vector_type(16))) _Float16 v16h;
typedef __attribute__((ext_vector_type(8)))  float    v8f;

// ---------------------------------------------------------------------------
// WMMA GEMM: C[M,Np] = epilogue(A[M,*lda] @ Wp + bias) via
// V_WMMA_F32_16X16X32_F16 (wave32). One wave computes a 16x32 C strip
// (two 16x16 tiles sharing the A fragment). Weights are pre-packed f16,
// zero-padded so NO bounds checks are needed in the K loop:
//   packed[((k/32)*Np + n)*32 + (k%32)]  -> lane's B frag is 16 contiguous f16.
// A rows may be read past the logical K (garbage x 0 = 0). M % 16 == 0 always.
// act: 0=none 1=gelu(exact) 2=sigmoid 3=clip[-10,2]; accum: 0 store, 1 +=.
// ---------------------------------------------------------------------------
__global__ __launch_bounds__(32)
void wmma_gemm_kernel(const float* __restrict__ A, int lda,
                      const _Float16* __restrict__ Wp,
                      const float* __restrict__ bias, float* __restrict__ C,
                      int nkc, int Np, int Nout, int act, int accum)
{
    const int lane = threadIdx.x;
    const int half = lane >> 4;
    const int l15  = lane & 15;
    const int m_a  = blockIdx.y * 16 + l15;      // A row this lane streams
    const int n0   = blockIdx.x * 32 + l15;      // C column, tile 0
    const int n1   = n0 + 16;                    // C column, tile 1
    const float* __restrict__ arow = A + (size_t)m_a * lda;

    v8f c0 = {0.f,0.f,0.f,0.f,0.f,0.f,0.f,0.f};
    v8f c1 = {0.f,0.f,0.f,0.f,0.f,0.f,0.f,0.f};

    for (int kc = 0; kc < nkc; ++kc) {
        const int k0 = kc * 32;
        // A fragment: 16 contiguous floats (k = 16*half + s), 16B aligned (lda % 4 == 0)
        const float4* ap0 = reinterpret_cast<const float4*>(arow + k0 + 16*half);
        const float4* ap1 = reinterpret_cast<const float4*>(arow + k0 + 16*half + 8);
        float4 a0 = ap0[0], a1 = ap0[1];
        float4 a2 = ap1[0], a3 = ap1[1];
        v16h a;
        a[0]=(_Float16)a0.x;  a[1]=(_Float16)a0.y;  a[2]=(_Float16)a0.z;  a[3]=(_Float16)a0.w;
        a[4]=(_Float16)a1.x;  a[5]=(_Float16)a1.y;  a[6]=(_Float16)a1.z;  a[7]=(_Float16)a1.w;
        a[8]=(_Float16)a2.x;  a[9]=(_Float16)a2.y;  a[10]=(_Float16)a2.z; a[11]=(_Float16)a2.w;
        a[12]=(_Float16)a3.x; a[13]=(_Float16)a3.y; a[14]=(_Float16)a3.z; a[15]=(_Float16)a3.w;
        // B fragments: 16 contiguous f16 per lane (32B aligned)
        const v16h b0 = *reinterpret_cast<const v16h*>(Wp + (((size_t)kc*Np + n0)*32) + 16*half);
        const v16h b1 = *reinterpret_cast<const v16h*>(Wp + (((size_t)kc*Np + n1)*32) + 16*half);
        c0 = __builtin_amdgcn_wmma_f32_16x16x32_f16(false, a, false, b0, (short)0, c0, false, false);
        c1 = __builtin_amdgcn_wmma_f32_16x16x32_f16(false, a, false, b1, (short)0, c1, false, false);
    }

    const int mbase = blockIdx.y * 16 + 8*half;  // C layout: VGPR r -> row r + 8*half
    if (n0 < Nout) {
        float bb = bias ? bias[n0] : 0.0f;
#pragma unroll
        for (int r = 0; r < 8; ++r) {
            float v = c0[r] + bb;
            if (act == 1)      v = 0.5f * v * (1.0f + erff(v * 0.70710678118f));
            else if (act == 2) v = 1.0f / (1.0f + expf(-v));
            else if (act == 3) v = fminf(2.0f, fmaxf(-10.0f, v));
            size_t idx = (size_t)(mbase + r) * Np + n0;
            if (accum) C[idx] += v; else C[idx] = v;
        }
    }
    if (n1 < Nout) {
        float bb = bias ? bias[n1] : 0.0f;
#pragma unroll
        for (int r = 0; r < 8; ++r) {
            float v = c1[r] + bb;
            if (act == 1)      v = 0.5f * v * (1.0f + erff(v * 0.70710678118f));
            else if (act == 2) v = 1.0f / (1.0f + expf(-v));
            else if (act == 3) v = fminf(2.0f, fmaxf(-10.0f, v));
            size_t idx = (size_t)(mbase + r) * Np + n1;
            if (accum) C[idx] += v; else C[idx] = v;
        }
    }
}

// Pack W[Kt,Nt] (f32 row-major) -> f16, zero-padded to [Kp,Np], WMMA-B layout.
__global__ void pack_weight_kernel(const float* __restrict__ W, _Float16* __restrict__ out,
                                   int Kt, int Nt, int Kp, int Np)
{
    long idx = (long)blockIdx.x * blockDim.x + threadIdx.x;
    if (idx >= (long)Kp * Np) return;
    int n = (int)(idx % Np);
    int k = (int)(idx / Np);
    float v = (k < Kt && n < Nt) ? W[(size_t)k * Nt + n] : 0.0f;
    out[(((size_t)(k >> 5)) * Np + n) * 32 + (k & 31)] = (_Float16)v;
}

// ---------------------------------------------------------------------------
// Elementwise / helper kernels (B=2, P=6, T=24, L=32, H=64, CE=16, S=64 fixed)
// ---------------------------------------------------------------------------
__global__ void fill_kernel(float* p, float v, long n) {
    long i = (long)blockIdx.x * blockDim.x + threadIdx.x;
    if (i < n) p[i] = v;
}

// cat[r, 0:96] = [prefix[b,p,n,:D] | spatial[n,:64] | ce[b,:16] | 0...], r = b*N+n
__global__ void build_inf_input_kernel(const float* __restrict__ prefix,
                                       const float* __restrict__ spatial,
                                       const float* __restrict__ ce,
                                       float* __restrict__ out,
                                       int p, int N, int D)
{
    long total = 2L * N * 96;
    long idx = (long)blockIdx.x * blockDim.x + threadIdx.x;
    if (idx >= total) return;
    int c = (int)(idx % 96);
    long r = idx / 96;
    int b = (int)(r / N), n = (int)(r % N);
    float v;
    if (c < D)            v = prefix[(((size_t)b*6 + p)*N + n)*D + c];
    else if (c < D + 64)  v = spatial[(size_t)n*64 + (c - D)];
    else if (c < D + 80)  v = ce[b*16 + (c - D - 64)];
    else                  v = 0.0f;
    out[(size_t)r * 96 + c] = v;
}

// GRU gate update: h = (1-z)*tanh(in + r*hn) + z*h   (gi/gh stride 192)
__global__ void gru_gates_kernel(const float* __restrict__ gi,
                                 const float* __restrict__ gh,
                                 float* __restrict__ h, int R)
{
    long idx = (long)blockIdx.x * blockDim.x + threadIdx.x;
    if (idx >= (long)R * 64) return;
    int r = (int)(idx >> 6), j = (int)(idx & 63);
    const float* gir = gi + (size_t)r * 192;
    const float* ghr = gh + (size_t)r * 192;
    float rg = 1.0f / (1.0f + expf(-(gir[j] + ghr[j])));
    float zg = 1.0f / (1.0f + expf(-(gir[64 + j] + ghr[64 + j])));
    float ng = tanhf(gir[128 + j] + rg * ghr[128 + j]);
    h[idx] = (1.0f - zg) * ng + zg * h[idx];
}

__device__ __forceinline__ unsigned hash_u32(unsigned x) {
    x ^= x >> 16; x *= 0x7feb352dU; x ^= x >> 15; x *= 0x846ca68bU; x ^= x >> 16;
    return x;
}

// z = mu + eps * exp(0.5*lv) (deterministic hash Box-Muller stand-in)
__global__ void sample_z_kernel(const float* __restrict__ mu, const float* __restrict__ lv,
                                float* __restrict__ z, long n, unsigned seed)
{
    long i = (long)blockIdx.x * blockDim.x + threadIdx.x;
    if (i >= n) return;
    unsigned h1 = hash_u32((unsigned)(2*i) + seed * 0x9E3779B9U);
    unsigned h2 = hash_u32((unsigned)(2*i + 1) ^ (seed * 0x85EBCA6BU));
    float u1 = ((float)h1 + 1.0f) * (1.0f / 4294967296.0f);
    float u2 = (float)h2 * (1.0f / 4294967296.0f);
    float eps = sqrtf(-2.0f * logf(u1)) * cosf(6.28318530718f * u2);
    z[i] = mu[i] + eps * expf(0.5f * lv[i]);
}

__global__ void count_kernel(const int* __restrict__ edge, float* cnt, int E) {
    int e = blockIdx.x * blockDim.x + threadIdx.x;
    if (e < E) atomicAdd(&cnt[edge[E + e]], 1.0f);
}

// agg[b,dst,f] += src[b,src,f] for both batches (L=32 features per edge)
__global__ void scatter_add_kernel(const float* __restrict__ src, const int* __restrict__ edge,
                                   float* __restrict__ agg, int E, int Nsrc, int Ndst)
{
    long idx = (long)blockIdx.x * blockDim.x + threadIdx.x;
    if (idx >= (long)E * 32) return;
    int e = (int)(idx >> 5), f = (int)(idx & 31);
    int s = edge[e], d = edge[E + e];
    atomicAdd(&agg[((size_t)d) * 32 + f],          src[((size_t)s) * 32 + f]);
    atomicAdd(&agg[((size_t)(Ndst + d)) * 32 + f], src[((size_t)(Nsrc + s)) * 32 + f]);
}

__global__ void mean_div_kernel(const float* __restrict__ agg, const float* __restrict__ cnt,
                                float* __restrict__ mean, int Nd)
{
    long idx = (long)blockIdx.x * blockDim.x + threadIdx.x;
    if (idx >= 2L * Nd * 32) return;
    int n = (int)((idx >> 5) % Nd);
    float c = cnt[n]; c = c < 1.0f ? 1.0f : c;
    mean[idx] = agg[idx] / c;
}

// In-place LayerNorm over rows of 32, one wave32 per row
__global__ void layernorm32_kernel(float* __restrict__ x, const float* __restrict__ g,
                                   const float* __restrict__ b, int M)
{
    int row = blockIdx.x;
    if (row >= M) return;
    int lane = threadIdx.x;
    float v = x[(size_t)row * 32 + lane];
    float s = v;
    for (int m = 16; m > 0; m >>= 1) s += __shfl_xor(s, m, 32);
    float mean = s * (1.0f / 32.0f);
    float dv = (v - mean) * (v - mean);
    for (int m = 16; m > 0; m >>= 1) dv += __shfl_xor(dv, m, 32);
    float var = dv * (1.0f / 32.0f);
    x[(size_t)row * 32 + lane] = (v - mean) * rsqrtf(var + 1e-5f) * g[lane] + b[lane];
}

__global__ void accum2_kernel(float* ds, float* zc, const float* d, long n) {
    long i = (long)blockIdx.x * blockDim.x + threadIdx.x;
    if (i < n) { float t = d[i]; ds[i] += t; zc[i] += t; }
}

__global__ void step_init_kernel(const float* z, float* zc, float* ds, long n) {
    long i = (long)blockIdx.x * blockDim.x + threadIdx.x;
    if (i < n) { zc[i] = z[i]; ds[i] = 0.0f; }
}

// cat[r, 0:64] = [z[b,n,:32] | ce[b,:16] | u[b,t,n,:CU] | 0...]
__global__ void build_trans_input_kernel(const float* __restrict__ z, const float* __restrict__ ce,
                                         const float* __restrict__ u, float* __restrict__ out,
                                         int t, int N, int CU)
{
    long total = 2L * N * 64;
    long idx = (long)blockIdx.x * blockDim.x + threadIdx.x;
    if (idx >= total) return;
    int c = (int)(idx & 63);
    long r = idx >> 6;
    int b = (int)(r / N), n = (int)(r % N);
    float v;
    if (c < 32)           v = z[(size_t)r * 32 + c];
    else if (c < 48)      v = ce[b*16 + (c - 32)];
    else if (c < 48 + CU) v = u[(((size_t)b*24 + t)*N + n)*CU + (c - 48)];
    else                  v = 0.0f;
    out[(size_t)r * 64 + c] = v;
}

// out[r,:64] = [a[r,:32] | b[r,:32]]
__global__ void concat2_kernel(const float* __restrict__ a, const float* __restrict__ b,
                               float* __restrict__ out, int M)
{
    long idx = (long)blockIdx.x * blockDim.x + threadIdx.x;
    if (idx >= (long)M * 64) return;
    int c = (int)(idx & 63); long r = idx >> 6;
    out[idx] = (c < 32) ? a[r*32 + c] : b[r*32 + (c - 32)];
}

// cat[r,:96] = [z[b,n,:32] | spatial[n,:64]]
__global__ void build_dec_input_kernel(const float* __restrict__ z, const float* __restrict__ sp,
                                       float* __restrict__ out, int N)
{
    long idx = (long)blockIdx.x * blockDim.x + threadIdx.x;
    if (idx >= 2L * N * 96) return;
    int c = (int)(idx % 96);
    long r = idx / 96;
    int n = (int)(r % N);
    out[(size_t)r * 96 + c] = (c < 32) ? z[r*32 + c] : sp[(size_t)n*64 + (c - 32)];
}

__global__ void gate_update_kernel(float* z, const float* g, const float* d, long n) {
    long i = (long)blockIdx.x * blockDim.x + threadIdx.x;
    if (i < n) z[i] += g[i] * d[i];
}

// out[b, t, ofs+n, 0] = val[(b*N+n)*stride]; out layout (2, 24, 44000, 1)
__global__ void copy_out_kernel(const float* __restrict__ val, int stride,
                                float* __restrict__ out, int t, int N, int ofs)
{
    long idx = (long)blockIdx.x * blockDim.x + threadIdx.x;
    if (idx >= 2L * N) return;
    int b = (int)(idx / N), n = (int)(idx % N);
    out[((size_t)b*24 + t) * 44000 + ofs + n] = val[idx * (size_t)stride];
}

// ---------------------------------------------------------------------------
// Host orchestration
// ---------------------------------------------------------------------------
extern "C" void kernel_launch(void* const* d_in, const int* in_sizes, int n_in,
                              void* d_out, int out_size, void* d_ws, size_t ws_size,
                              hipStream_t stream)
{
    (void)in_sizes; (void)n_in; (void)out_size; (void)ws_size;
    auto F = [&](int i) { return (const float*)d_in[i]; };
    auto I = [&](int i) { return (const int*)d_in[i]; };

    // ---- input unpack (jax pytree sorted-key flattening) -------------------
    const float* prefix1 = F(0);  const float* prefix2 = F(1);
    const float* sp1 = F(2);      const float* sp2 = F(3);
    const float* ce  = F(4);
    const float* u1  = F(5);      const float* u2  = F(6);
    const int* e_pipe = I(99);   const int Epipe = 8000;
    const int* e_surf = I(100);  const int Esurf = 160000;
    const int* e_ct   = I(101);  const int Ect   = 4000;
    const int* e_cf   = I(102);  const int Ecf   = 4000;
    float* out = (float*)d_out;

    // ---- workspace arena ---------------------------------------------------
    float* WSB = (float*)d_ws;
    size_t off = 0;
    auto alloc = [&](size_t n) { float* p = WSB + off; off += n; return p; };
    float* catb = alloc(80000ULL * 96 + 64);
    float* hidA = alloc(80000ULL * 64 + 64);
    float* gi   = alloc(80000ULL * 192);   // also start of transition overlay
    float* gh   = alloc(80000ULL * 192);   (void)gh;
    float* h0   = alloc(80000ULL * 64 + 64);
    float* h1   = alloc(80000ULL * 64 + 64);
    float* mu   = alloc(80000ULL * 32 + 64);   // reused as decoder output (stride 32)
    float* lv   = alloc(80000ULL * 32 + 64);
    float* z1   = alloc(2ULL * 4000 * 32 + 64);
    float* z2   = alloc(2ULL * 40000 * 32 + 64);
    float* cntp = alloc(4000);
    float* cnts = alloc(40000);
    float* cntct= alloc(40000);
    float* cntcf= alloc(4032);
    _Float16* hpack = (_Float16*)alloc(128000);   // packed f16 weights (~0.4 MB)
    // transition-phase buffers overlaid on gi+gh (30.72M floats, 23.8M used)
    size_t oo = 0;
    auto oalloc = [&](size_t n) { float* p = gi + oo; oo += n; return p; };
    float* z1c   = oalloc(2ULL*4000*32 + 64);
    float* z2c   = oalloc(2ULL*40000*32 + 64);
    float* ds1   = oalloc(2ULL*4000*32 + 64);
    float* ds2   = oalloc(2ULL*40000*32 + 64);
    float* d1b   = oalloc(2ULL*4000*32 + 64);
    float* d2b   = oalloc(2ULL*40000*32 + 64);
    float* aggb  = oalloc(2ULL*40000*32 + 64);
    float* meanb = oalloc(2ULL*40000*32 + 64);
    float* outsum= oalloc(2ULL*40000*64 + 64);
    float* hidB  = oalloc(2ULL*40000*64 + 64);

    auto ew = [](long n) { return dim3((unsigned)((n + 255) / 256)); };
    auto fill = [&](float* p, float v, long n) {
        fill_kernel<<<ew(n), 256, 0, stream>>>(p, v, n);
    };
    // gemm(A, lda, packedW, bias, C, M, nkc=K/32, Np, Nout, act, accum)
    auto gemm = [&](const float* A, int lda, const _Float16* Wp, const float* bias, float* C,
                    int M, int nkc, int Np, int Nout, int act, int accum) {
        dim3 g(Np / 32, M / 16);
        wmma_gemm_kernel<<<g, dim3(32), 0, stream>>>(A, lda, Wp, bias, C, nkc, Np, Nout, act, accum);
    };
    // pack a weight once per launch
    size_t hoff = 0;
    auto pack = [&](const float* Wsrc, int Kt, int Nt, int Kp, int Np) {
        _Float16* p = hpack + hoff; hoff += (size_t)Kp * Np;
        pack_weight_kernel<<<ew((long)Kp*Np), 256, 0, stream>>>(Wsrc, p, Kt, Nt, Kp, Np);
        return (const _Float16*)p;
    };

    // ---- pack all weights --------------------------------------------------
    struct InfW { const _Float16 *in_w,*Wih0,*Whh0,*Wih1,*Whh1,*mu_w,*lv_w;
                  const float *in_b,*bih0,*bhh0,*bih1,*bhh1,*mu_b,*lv_b; };
    auto pack_inf = [&](int base) {
        InfW w;
        w.Whh0 = pack(F(base+0), 64, 192, 64, 192);
        w.Whh1 = pack(F(base+1), 64, 192, 64, 192);
        w.Wih0 = pack(F(base+2), 64, 192, 64, 192);
        w.Wih1 = pack(F(base+3), 64, 192, 64, 192);
        w.bhh0 = F(base+4); w.bhh1 = F(base+5); w.bih0 = F(base+6); w.bih1 = F(base+7);
        w.in_b = F(base+8);
        int Din = (base == 19) ? 84 : 83;
        w.in_w = pack(F(base+9), Din, 64, 96, 64);
        w.lv_b = F(base+10); w.lv_w = pack(F(base+11), 64, 32, 64, 32);
        w.mu_b = F(base+12); w.mu_w = pack(F(base+13), 64, 32, 64, 32);
        return w;
    };
    InfW inf1 = pack_inf(19), inf2 = pack_inf(33);

    struct GnnW { const _Float16 *cf_wl,*cf_wr,*ct_wl,*ct_wr,*pipe_wl,*pipe_wr,
                                 *surf_wl,*surf_wr,*proj1_w,*proj2_w;
                  const float *cf_bl,*ct_bl,*pipe_bl,*surf_bl,
                              *proj1_b,*proj2_b,*ln1_g,*ln1_b,*ln2_g,*ln2_b; };
    auto pack_gnn = [&](int gb) {
        GnnW g;
        g.cf_bl = F(gb+0);   g.cf_wl  = pack(F(gb+1), 32, 64, 32, 64);
        g.cf_wr = pack(F(gb+2), 32, 64, 32, 64);
        g.ct_bl = F(gb+3);   g.ct_wl  = pack(F(gb+4), 32, 64, 32, 64);
        g.ct_wr = pack(F(gb+5), 32, 64, 32, 64);
        g.ln1_b = F(gb+6);   g.ln1_g = F(gb+7);  g.ln2_b = F(gb+8);  g.ln2_g = F(gb+9);
        g.pipe_bl = F(gb+10); g.pipe_wl = pack(F(gb+11), 32, 64, 32, 64);
        g.pipe_wr = pack(F(gb+12), 32, 64, 32, 64);
        g.proj1_b = F(gb+13); g.proj1_w = pack(F(gb+14), 64, 32, 64, 32);
        g.proj2_b = F(gb+15); g.proj2_w = pack(F(gb+16), 64, 32, 64, 32);
        g.surf_bl = F(gb+17); g.surf_wl = pack(F(gb+18), 32, 64, 32, 64);
        g.surf_wr = pack(F(gb+19), 32, 64, 32, 64);
        return g;
    };
    GnnW gnn0 = pack_gnn(51), gnn1 = pack_gnn(71);

    const _Float16* g1_wp  = pack(F(48), 64, 32, 64, 32);  const float* g1_b = F(47);
    const _Float16* g2_wp  = pack(F(50), 64, 32, 64, 32);  const float* g2_b = F(49);
    const _Float16* t1_w1p = pack(F(93), 50, 64, 64, 64);  const float* t1_b1 = F(91);
    const _Float16* t1_w2p = pack(F(94), 64, 32, 64, 32);  const float* t1_b2 = F(92);
    const _Float16* t2_w1p = pack(F(97), 49, 64, 64, 64);  const float* t2_b1 = F(95);
    const _Float16* t2_w2p = pack(F(98), 64, 32, 64, 32);  const float* t2_b2 = F(96);

    struct DecW { const _Float16 *w1,*w2,*w3; const float *b1,*b2,*b3; };
    auto pack_dec = [&](int base) {
        DecW d;
        d.b1 = F(base+0); d.b2 = F(base+1); d.b3 = F(base+2);
        d.w1 = pack(F(base+3), 96, 64, 96, 64);
        d.w2 = pack(F(base+4), 64, 64, 64, 64);
        d.w3 = pack(F(base+5), 64, 1, 64, 32);     // N padded 1 -> 32
        return d;
    };
    DecW dec1 = pack_dec(7), dec2 = pack_dec(13);

    // ---- edge degree counts (constant across timesteps) --------------------
    fill(cntp, 0.f, 4000);  count_kernel<<<ew(Epipe),256,0,stream>>>(e_pipe, cntp, Epipe);
    fill(cnts, 0.f, 40000); count_kernel<<<ew(Esurf),256,0,stream>>>(e_surf, cnts, Esurf);
    fill(cntct,0.f, 40000); count_kernel<<<ew(Ect),  256,0,stream>>>(e_ct,   cntct, Ect);
    fill(cntcf,0.f, 4000);  count_kernel<<<ew(Ecf),  256,0,stream>>>(e_cf,   cntcf, Ecf);

    // ---- inference nets ----------------------------------------------------
    auto run_inf = [&](const float* prefix, const float* spatial, const InfW& w,
                       int N, int D, float* zout, unsigned seed) {
        int R = 2 * N;
        fill(h0, 0.f, (long)R * 64);
        fill(h1, 0.f, (long)R * 64);
        for (int p = 0; p < 6; ++p) {
            build_inf_input_kernel<<<ew((long)R*96),256,0,stream>>>(prefix, spatial, ce, catb, p, N, D);
            gemm(catb, 96, w.in_w, w.in_b, hidA, R, 3, 64, 64, 0, 0);
            gemm(hidA, 64, w.Wih0, w.bih0, gi, R, 2, 192, 192, 0, 0);
            gemm(h0,   64, w.Whh0, w.bhh0, gh, R, 2, 192, 192, 0, 0);
            gru_gates_kernel<<<ew((long)R*64),256,0,stream>>>(gi, gh, h0, R);
            gemm(h0, 64, w.Wih1, w.bih1, gi, R, 2, 192, 192, 0, 0);
            gemm(h1, 64, w.Whh1, w.bhh1, gh, R, 2, 192, 192, 0, 0);
            gru_gates_kernel<<<ew((long)R*64),256,0,stream>>>(gi, gh, h1, R);
        }
        gemm(h1, 64, w.mu_w, w.mu_b, mu, R, 2, 32, 32, 0, 0);
        gemm(h1, 64, w.lv_w, w.lv_b, lv, R, 2, 32, 32, 3, 0);   // clip[-10,2]
        sample_z_kernel<<<ew((long)R*32),256,0,stream>>>(mu, lv, zout, (long)R*32, seed);
    };
    run_inf(prefix1, sp1, inf1, 4000,  4, z1, 42u);
    run_inf(prefix2, sp2, inf2, 40000, 3, z2, 43u);

    // ---- transition + decode loop ------------------------------------------
    auto sage = [&](const float* srcz, int Nsrc, int Ndst, const int* edge, int E,
                    const float* cnt, const _Float16* wl, const float* bl,
                    const _Float16* wr, const float* xdst, float* outb, bool first) {
        long nagg = 2L * Ndst * 32;
        fill(aggb, 0.f, nagg);
        scatter_add_kernel<<<ew((long)E*32),256,0,stream>>>(srcz, edge, aggb, E, Nsrc, Ndst);
        mean_div_kernel<<<ew(nagg),256,0,stream>>>(aggb, cnt, meanb, Ndst);
        gemm(meanb, 32, wl, bl, outb, 2*Ndst, 1, 64, 64, 0, first ? 0 : 1);
        gemm(xdst,  32, wr, nullptr, outb, 2*Ndst, 1, 64, 64, 0, 1);
    };

    for (int t = 0; t < 24; ++t) {
        step_init_kernel<<<ew(256000),256,0,stream>>>(z1, z1c, ds1, 256000);
        step_init_kernel<<<ew(2560000),256,0,stream>>>(z2, z2c, ds2, 2560000);

        for (int blk = 0; blk < 2; ++blk) {
            const GnnW& g = (blk == 0) ? gnn0 : gnn1;
            // out1 = sage(z1c->z1c, pipe) + sage(z2c->z1c, cf)
            sage(z1c, 4000,  4000, e_pipe, Epipe, cntp,  g.pipe_wl, g.pipe_bl, g.pipe_wr, z1c, outsum, true);
            sage(z2c, 40000, 4000, e_cf,   Ecf,   cntcf, g.cf_wl,   g.cf_bl,   g.cf_wr,   z1c, outsum, false);
            gemm(outsum, 64, g.proj1_w, g.proj1_b, d1b, 8000, 2, 32, 32, 0, 0);
            layernorm32_kernel<<<8000,32,0,stream>>>(d1b, g.ln1_g, g.ln1_b, 8000);
            // out2 = sage(z2c->z2c, surf) + sage(z1c->z2c, ct)
            sage(z2c, 40000, 40000, e_surf, Esurf, cnts,  g.surf_wl, g.surf_bl, g.surf_wr, z2c, outsum, true);
            sage(z1c, 4000,  40000, e_ct,   Ect,   cntct, g.ct_wl,   g.ct_bl,   g.ct_wr,   z2c, outsum, false);
            gemm(outsum, 64, g.proj2_w, g.proj2_b, d2b, 80000, 2, 32, 32, 0, 0);
            layernorm32_kernel<<<80000,32,0,stream>>>(d2b, g.ln2_g, g.ln2_b, 80000);
            accum2_kernel<<<ew(256000),256,0,stream>>>(ds1, z1c, d1b, 256000);
            accum2_kernel<<<ew(2560000),256,0,stream>>>(ds2, z2c, d2b, 2560000);
        }

        // dt1 = MLP([z1|ce|u1_t]); d1 = ds1 + dt1; gate; z1 += g1*d1
        build_trans_input_kernel<<<ew(8000L*64),256,0,stream>>>(z1, ce, u1, catb, t, 4000, 2);
        gemm(catb, 64, t1_w1p, t1_b1, hidA, 8000, 2, 64, 64, 1, 0);
        gemm(hidA, 64, t1_w2p, t1_b2, ds1,  8000, 2, 32, 32, 0, 1);
        concat2_kernel<<<ew(8000L*64),256,0,stream>>>(z1, ds1, catb, 8000);
        gemm(catb, 64, g1_wp, g1_b, d1b, 8000, 2, 32, 32, 2, 0);
        gate_update_kernel<<<ew(256000),256,0,stream>>>(z1, d1b, ds1, 256000);

        build_trans_input_kernel<<<ew(80000L*64),256,0,stream>>>(z2, ce, u2, catb, t, 40000, 1);
        gemm(catb, 64, t2_w1p, t2_b1, hidA, 80000, 2, 64, 64, 1, 0);
        gemm(hidA, 64, t2_w2p, t2_b2, ds2,  80000, 2, 32, 32, 0, 1);
        concat2_kernel<<<ew(80000L*64),256,0,stream>>>(z2, ds2, catb, 80000);
        gemm(catb, 64, g2_wp, g2_b, d2b, 80000, 2, 32, 32, 2, 0);
        gate_update_kernel<<<ew(2560000),256,0,stream>>>(z2, d2b, ds2, 2560000);

        // decode 1 -> out[:, t, 0:4000]
        build_dec_input_kernel<<<ew(8000L*96),256,0,stream>>>(z1, sp1, catb, 4000);
        gemm(catb, 96, dec1.w1, dec1.b1, hidA, 8000, 3, 64, 64, 1, 0);
        gemm(hidA, 64, dec1.w2, dec1.b2, hidB, 8000, 2, 64, 64, 1, 0);
        gemm(hidB, 64, dec1.w3, dec1.b3, mu,   8000, 2, 32, 1,  0, 0);
        copy_out_kernel<<<ew(8000),256,0,stream>>>(mu, 32, out, t, 4000, 0);
        // decode 2 -> out[:, t, 4000:44000]
        build_dec_input_kernel<<<ew(80000L*96),256,0,stream>>>(z2, sp2, catb, 40000);
        gemm(catb, 96, dec2.w1, dec2.b1, hidA, 80000, 3, 64, 64, 1, 0);
        gemm(hidA, 64, dec2.w2, dec2.b2, hidB, 80000, 2, 64, 64, 1, 0);
        gemm(hidB, 64, dec2.w3, dec2.b3, mu,   80000, 2, 32, 1,  0, 0);
        copy_out_kernel<<<ew(80000),256,0,stream>>>(mu, 32, out, t, 40000, 4000);
    }
}
